// RotatedModule_17712445129289
// MI455X (gfx1250) — matrix operand order repaired and checked by reading, not compiled, not run on hardware
//
#include <hip/hip_runtime.h>
#include <hip/hip_bf16.h>
#include <math.h>

// ---------------------------------------------------------------------------
// out[m, o] = sum_d FWHT(x)[m, d] * W[o, d] + b[o]
// Hadamard H is symmetric => (Hx)·W_o = x·(H W_o). So:
//   1) Wr = FWHT(W rows)    (async global<->LDS copies, ASYNCcnt tracked)
//   2) out = x @ Wr^T + b   (bf16 WMMA GEMM, hi/lo split ~f32 accuracy,
//                            register-pipelined global loads)
// ---------------------------------------------------------------------------

typedef __attribute__((ext_vector_type(16))) __bf16 v16bf;
typedef __attribute__((ext_vector_type(4)))  __bf16 v4bf;
typedef __attribute__((ext_vector_type(8)))  float  v8f;

// ---------------------------------------------------------------------------
// Kernel 1: normalized FWHT along rows of W (length K, power of two).
// Row copied in with GLOBAL_LOAD_ASYNC_TO_LDS_B128 and out with
// GLOBAL_STORE_ASYNC_FROM_LDS_B128 (dynamic LDS starts at offset 0 here).
// ---------------------------------------------------------------------------
__global__ __launch_bounds__(256) void fwht_rows_kernel(
    const float* __restrict__ W, float* __restrict__ Wr, int K, float scale) {
  extern __shared__ float row[];
  const int t = threadIdx.x;
  const int r = blockIdx.x;
  const float* src = W + (size_t)r * K;
  float* dst = Wr + (size_t)r * K;

  // async global -> LDS, 16B per lane per trip
  for (int i = t * 4; i < K; i += 256 * 4) {
    const unsigned lds_off = (unsigned)(i * 4);   // bytes into dynamic LDS
    const float* ga = src + i;
    asm volatile("global_load_async_to_lds_b128 %0, %1, off"
                 :: "v"(lds_off), "v"(ga) : "memory");
  }
  asm volatile("s_wait_asynccnt 0x0" ::: "memory");
  __syncthreads();

  for (int h = 1; h < K; h <<= 1) {
    for (int s = t; s < (K >> 1); s += 256) {
      const int grp = s / h;
      const int off = s - grp * h;
      const int i0  = grp * 2 * h + off;
      const int i1  = i0 + h;
      const float a = row[i0];
      const float c = row[i1];
      row[i0] = a + c;
      row[i1] = a - c;
    }
    __syncthreads();
  }

  // scale exactly the 16B chunks this thread will store, then async LDS->global
  for (int i = t * 4; i < K; i += 256 * 4) {
    row[i + 0] *= scale;
    row[i + 1] *= scale;
    row[i + 2] *= scale;
    row[i + 3] *= scale;
    const unsigned lds_off = (unsigned)(i * 4);
    float* gd = dst + i;
    asm volatile("global_store_async_from_lds_b128 %0, %1, off"
                 :: "v"(gd), "v"(lds_off) : "memory");
  }
  asm volatile("s_wait_asynccnt 0x0" ::: "memory");
}

// ---------------------------------------------------------------------------
// Kernel 2: GEMM  out[M,N] = X[M,K] * Wr[N,K]^T + bias[N]
// Workgroup tile 128x128, BK=32, 8 waves (wave32) in 2x4 grid, wave tile 64x32.
// LDS holds bf16 hi/lo tiles; fragments are raw ds_load_b128; next K-strip's
// global loads are issued before compute (register double-buffer).
// Assumes M%128==0, N%128==0, K%32==0 (8192/4096/4096 here).
// ---------------------------------------------------------------------------
#define BM 128
#define BN 128
#define BK 32
#define TS 40   // bf16 row stride: 80 bytes (16B-aligned, bank-skewed)

__device__ __forceinline__ void split4(const float4 v, v4bf& hi, v4bf& lo) {
  const __bf16 h0 = (__bf16)v.x, h1 = (__bf16)v.y,
               h2 = (__bf16)v.z, h3 = (__bf16)v.w;
  hi[0] = h0; hi[1] = h1; hi[2] = h2; hi[3] = h3;
  lo[0] = (__bf16)(v.x - (float)h0);
  lo[1] = (__bf16)(v.y - (float)h1);
  lo[2] = (__bf16)(v.z - (float)h2);
  lo[3] = (__bf16)(v.w - (float)h3);
}

__global__ __launch_bounds__(256) void gemm_fwht_kernel(
    const float* __restrict__ X, const float* __restrict__ Wr,
    const float* __restrict__ bias, float* __restrict__ out,
    int M, int N, int K) {
  __shared__ __bf16 Ah[BM][TS];   // A tile hi: 128 (M) x 32 (K)
  __shared__ __bf16 Al[BM][TS];   // A tile lo
  __shared__ __bf16 Bh[BN][TS];   // B tile hi: 128 (N) x 32 (K)
  __shared__ __bf16 Bl[BN][TS];   // B tile lo

  const int t       = threadIdx.x;
  const int lane    = t & 31;
  const int wave    = t >> 5;      // 0..7
  const int wave_m  = wave >> 2;   // 0..1  -> 64-row M slab
  const int wave_n  = wave & 3;    // 0..3  -> 32-col N slab
  const int ln16    = lane & 15;
  const bool hiHalf = lane >= 16;

  const int n0 = blockIdx.x * BN;
  const int m0 = blockIdx.y * BM;

  // per-thread staging coordinates (4 float4 slots per tile)
  int srow[4], scol[4];
#pragma unroll
  for (int p = 0; p < 4; ++p) {
    const int idx = t + p * 256;
    srow[p] = idx >> 3;            // 8 float4 per 32-float row
    scol[p] = (idx & 7) << 2;
  }

  v8f acc[4][2];
#pragma unroll
  for (int mi = 0; mi < 4; ++mi)
#pragma unroll
    for (int ni = 0; ni < 2; ++ni) acc[mi][ni] = (v8f)0.0f;

  const int kTiles = K / BK;

  // prologue: load K-strip 0 into registers
  float4 ca[4], cb[4];
#pragma unroll
  for (int p = 0; p < 4; ++p) {
    ca[p] = *(const float4*)(X  + (size_t)(m0 + srow[p]) * K + scol[p]);
    cb[p] = *(const float4*)(Wr + (size_t)(n0 + srow[p]) * K + scol[p]);
  }

  for (int kt = 0; kt < kTiles; ++kt) {
    // ---- split once, store bf16 hi/lo tiles to LDS ----
#pragma unroll
    for (int p = 0; p < 4; ++p) {
      v4bf h, l;
      split4(ca[p], h, l);
      *(v4bf*)&Ah[srow[p]][scol[p]] = h;
      *(v4bf*)&Al[srow[p]][scol[p]] = l;
      split4(cb[p], h, l);
      *(v4bf*)&Bh[srow[p]][scol[p]] = h;
      *(v4bf*)&Bl[srow[p]][scol[p]] = l;
    }
    __syncthreads();

    // ---- issue next K-strip's global loads; latency hides under WMMAs ----
    float4 na[4], nb[4];
    if (kt + 1 < kTiles) {
      const int k1 = (kt + 1) * BK;
#pragma unroll
      for (int p = 0; p < 4; ++p) {
        na[p] = *(const float4*)(X  + (size_t)(m0 + srow[p]) * K + k1 + scol[p]);
        nb[p] = *(const float4*)(Wr + (size_t)(n0 + srow[p]) * K + k1 + scol[p]);
      }
    }

    // ---- B fragments (32x16 K x N): lanes 0-15 K=0..15, lanes 16-31 K=16..31,
    //      N = lane&15. 16 consecutive bf16 => two ds_load_b128. ----
    v16bf b_hi[2], b_lo[2];
#pragma unroll
    for (int ni = 0; ni < 2; ++ni) {
      const int nrow = wave_n * 32 + ni * 16 + ln16;
      const int kb   = hiHalf ? 16 : 0;
      ((uint4*)&b_hi[ni])[0] = *(const uint4*)&Bh[nrow][kb];
      ((uint4*)&b_hi[ni])[1] = *(const uint4*)&Bh[nrow][kb + 8];
      ((uint4*)&b_lo[ni])[0] = *(const uint4*)&Bl[nrow][kb];
      ((uint4*)&b_lo[ni])[1] = *(const uint4*)&Bl[nrow][kb + 8];
    }

    // ---- per M-tile: A fragment (16x32 M x K): lanes 0-15 K runs {0..7,8..15},
    //      lanes 16-31 {16..23,24..31}; each run = one ds_load_b128. ----
#pragma unroll
    for (int mi = 0; mi < 4; ++mi) {
      const int mrow = wave_m * 64 + mi * 16 + ln16;
      const int kb   = hiHalf ? 16 : 0;
      v16bf a_hi, a_lo;
      ((uint4*)&a_hi)[0] = *(const uint4*)&Ah[mrow][kb];
      ((uint4*)&a_hi)[1] = *(const uint4*)&Ah[mrow][kb + 8];
      ((uint4*)&a_lo)[0] = *(const uint4*)&Al[mrow][kb];
      ((uint4*)&a_lo)[1] = *(const uint4*)&Al[mrow][kb + 8];

#pragma unroll
      for (int ni = 0; ni < 2; ++ni) {
        acc[mi][ni] = __builtin_amdgcn_wmma_f32_16x16x32_bf16(
            false, a_hi, false, b_hi[ni], (short)0, acc[mi][ni], false, false);
        acc[mi][ni] = __builtin_amdgcn_wmma_f32_16x16x32_bf16(
            false, a_hi, false, b_lo[ni], (short)0, acc[mi][ni], false, false);
        acc[mi][ni] = __builtin_amdgcn_wmma_f32_16x16x32_bf16(
            false, a_lo, false, b_hi[ni], (short)0, acc[mi][ni], false, false);
      }
    }
    __syncthreads();

    if (kt + 1 < kTiles) {
#pragma unroll
      for (int p = 0; p < 4; ++p) { ca[p] = na[p]; cb[p] = nb[p]; }
    }
  }

  // ---- epilogue: f32 C/D 16x16 layout: lane holds N=lane&15,
  //      rows M = (lane<16 ? 0..7 : 8..15) across 8 accumulator VGPRs ----
#pragma unroll
  for (int ni = 0; ni < 2; ++ni) {
    const int ng = n0 + wave_n * 32 + ni * 16 + ln16;
    const float bv = bias[ng];
#pragma unroll
    for (int mi = 0; mi < 4; ++mi) {
      const int mbase = m0 + wave_m * 64 + mi * 16 + (hiHalf ? 8 : 0);
#pragma unroll
      for (int v = 0; v < 8; ++v) {
        out[(size_t)(mbase + v) * N + ng] = acc[mi][ni][v] + bv;
      }
    }
  }
}

// ---------------------------------------------------------------------------
extern "C" void kernel_launch(void* const* d_in, const int* in_sizes, int n_in,
                              void* d_out, int out_size, void* d_ws, size_t ws_size,
                              hipStream_t stream) {
  const float* x  = (const float*)d_in[0];
  const float* W  = (const float*)d_in[1];
  const float* b  = (const float*)d_in[2];
  float* out = (float*)d_out;

  const int D_OUT = in_sizes[2];                               // 4096
  const int D_IN  = (int)((long long)in_sizes[1] / D_OUT);     // 4096
  const int Mrows = (int)((long long)in_sizes[0] / D_IN);      // 8192

  float* Wr = (float*)d_ws;  // D_OUT * D_IN f32 = 64 MiB scratch

  const float scale = 1.0f / sqrtf((float)D_IN);
  fwht_rows_kernel<<<dim3(D_OUT), dim3(256), D_IN * sizeof(float), stream>>>(
      W, Wr, D_IN, scale);

  dim3 grid(D_OUT / BN, Mrows / BM);
  gemm_fwht_kernel<<<grid, dim3(256), 0, stream>>>(x, Wr, b, out,
                                                   Mrows, D_OUT, D_IN);
}
